// EncoderDecoderV3_38079180046976
// MI455X (gfx1250) — matrix operand order repaired and checked by reading, not compiled, not run on hardware
//
#include <hip/hip_runtime.h>
#include <stdint.h>

typedef unsigned short ushort_t;
typedef __bf16 v16bf __attribute__((ext_vector_type(16)));
typedef float  v8f   __attribute__((ext_vector_type(8)));

#define B_   512
#define FP_  4096
#define ENC_ 64
#define H_   512
#define V_   42
#define SEQ_ 128
#define G_   (3 * H_)

// gfx1250 async copy (global -> LDS, ASYNCcnt-tracked) availability.
// Probe-confirmed signature: (AS1 int4* gsrc, AS3 int4* ldst, imm offset, imm cpol)
#if defined(__gfx1250__) && __has_builtin(__builtin_amdgcn_global_load_async_to_lds_b128)
#define ASYNC_LDS 1
typedef int v4i_t __attribute__((vector_size(16)));
typedef __attribute__((address_space(1))) v4i_t as1_v4i;
typedef __attribute__((address_space(3))) v4i_t as3_v4i;
#else
#define ASYNC_LDS 0
#endif

__device__ __forceinline__ void wait_asynccnt_3() {
#if ASYNC_LDS
#if __has_builtin(__builtin_amdgcn_s_wait_asynccnt)
  __builtin_amdgcn_s_wait_asynccnt(3);
#else
  asm volatile("s_wait_asynccnt 0x3" ::: "memory");
#endif
#endif
}
__device__ __forceinline__ void wait_asynccnt_0() {
#if ASYNC_LDS
#if __has_builtin(__builtin_amdgcn_s_wait_asynccnt)
  __builtin_amdgcn_s_wait_asynccnt(0);
#else
  asm volatile("s_wait_asynccnt 0x0" ::: "memory");
#endif
#endif
}

// ---------------- helpers ----------------
__device__ __forceinline__ ushort_t f2bf(float f) {
  uint32_t u = __float_as_uint(f);
  u += 0x7fffu + ((u >> 16) & 1u);   // round-to-nearest-even
  return (ushort_t)(u >> 16);
}
__device__ __forceinline__ float sigm(float x) { return 1.0f / (1.0f + expf(-x)); }

// ---------------- generic bf16 WMMA GEMM ----------------
// C[M,ldc] (f32, optional) / Cbf[M,N] (bf16, optional) = act(A[M,K] * W[N,K]^T + bias[N])
// Block tile 64(M) x 128(N) x 32(K); 8 wave32s; each wave -> 32x32 via 2x2 WMMA tiles.
#define BM 64
#define BN 128
#define BK 32
#define LDA_S 40   // padded LDS stride (ushorts): 80B = 20 banks, conflict-free
#define LDB_S 40

// per-K-step fragment loads + 4 WMMAs (ISA VGPR layouts)
__device__ __forceinline__ void frag_mma(const ushort_t* __restrict__ sA,
                                         const ushort_t* __restrict__ sB,
                                         int wm, int wn, int lm, int hh,
                                         v8f acc[2][2]) {
  union FragBF { uint32_t u[8]; v16bf v; } af[2], bfr[2];
#pragma unroll
  for (int mi = 0; mi < 2; ++mi) {
    const int m = wm * 32 + mi * 16 + lm;
#pragma unroll
    for (int vv = 0; vv < 8; ++vv) {
      const int k = 2 * vv + (vv >= 4 ? 8 : 0) + 8 * hh;   // A 16x32 bf16 layout
      af[mi].u[vv] = *(const uint32_t*)(&sA[m * LDA_S + k]);
    }
  }
#pragma unroll
  for (int ni = 0; ni < 2; ++ni) {
    const int n = wn * 32 + ni * 16 + lm;
#pragma unroll
    for (int vv = 0; vv < 8; ++vv) {
      const int k = 2 * vv + (vv >= 4 ? 8 : 0) + 8 * hh;   // B 32x16 bf16 layout
      bfr[ni].u[vv] = *(const uint32_t*)(&sB[n * LDB_S + k]);
    }
  }
#pragma unroll
  for (int mi = 0; mi < 2; ++mi)
#pragma unroll
    for (int ni = 0; ni < 2; ++ni)
      acc[mi][ni] = __builtin_amdgcn_wmma_f32_16x16x32_bf16(
          false, af[mi].v, false, bfr[ni].v, (short)0, acc[mi][ni], false, false);
}

__global__ __launch_bounds__(256)
void gemm_bf16_wmma(const ushort_t* __restrict__ A,
                    const ushort_t* __restrict__ W,
                    const float*    __restrict__ bias,
                    float*          __restrict__ C,
                    ushort_t*       __restrict__ Cbf,
                    int M, int N, int K, int ldc, int act) {
  __shared__ ushort_t sA[2][BM * LDA_S];
  __shared__ ushort_t sB[2][BN * LDB_S];

  const int tid  = threadIdx.x;
  const int lane = tid & 31;
  const int wave = tid >> 5;
  const int wm   = wave >> 2;      // 0..1
  const int wn   = wave & 3;       // 0..3
  const int lm   = lane & 15;
  const int hh   = lane >> 4;      // half-wave select
  const int m0   = blockIdx.y * BM;
  const int n0   = blockIdx.x * BN;

  const v8f vzero = {0.f, 0.f, 0.f, 0.f, 0.f, 0.f, 0.f, 0.f};
  v8f acc[2][2];
  acc[0][0] = vzero; acc[0][1] = vzero; acc[1][0] = vzero; acc[1][1] = vzero;

  const int ar = tid >> 2;          // A stage: 64 rows x 32 cols, one b128 per thread
  const int ac = (tid & 3) * 8;

#if ASYNC_LDS
  const bool fullN = (n0 + BN <= N);   // block-uniform
  if (fullN) {
    // ---- async double-buffered pipeline: tile k+1 streams cache->LDS while
    // ---- tile k runs through the WMMA pipe (no VGPR round-trip). 3 ops/thread.
    auto issue = [&](int buf, int k0) {
      __builtin_amdgcn_global_load_async_to_lds_b128(
          (as1_v4i*)(const void*)(A + (size_t)(m0 + ar) * K + k0 + ac),
          (as3_v4i*)&sA[buf][ar * LDA_S + ac], 0, 0);
#pragma unroll
      for (int i = 0; i < 2; ++i) {
        const int idx = tid + i * 256;
        const int br  = idx >> 2;
        const int bc  = (idx & 3) * 8;
        __builtin_amdgcn_global_load_async_to_lds_b128(
            (as1_v4i*)(const void*)(W + (size_t)(n0 + br) * K + k0 + bc),
            (as3_v4i*)&sB[buf][br * LDB_S + bc], 0, 0);
      }
    };
    const int nk = K / BK;
    issue(0, 0);
    for (int kk = 0; kk < nk; ++kk) {
      const int cur = kk & 1;
      if (kk + 1 < nk) {
        issue(cur ^ 1, (kk + 1) * BK);   // buf[cur^1] free since end-of-iter barrier of kk-1
        wait_asynccnt_3();               // in-order completion -> tile kk's 3 ops are done
      } else {
        wait_asynccnt_0();
      }
      __syncthreads();                   // all waves' copies for buf[cur] visible
      frag_mma(&sA[cur][0], &sB[cur][0], wm, wn, lm, hh, acc);
      __syncthreads();                   // done reading buf[cur] before it is refilled
    }
  } else
#endif
  {
    // ---- masked synchronous path (N tails: 42 / 64) ----
    for (int k0 = 0; k0 < K; k0 += BK) {
      {
        const uint4 va = *(const uint4*)(A + (size_t)(m0 + ar) * K + k0 + ac);
        *(uint4*)(&sA[0][ar * LDA_S + ac]) = va;
      }
      if (k0 + BK < K)
        __builtin_prefetch(A + (size_t)(m0 + ar) * K + k0 + BK + ac, 0, 1);
#pragma unroll
      for (int i = 0; i < 2; ++i) {
        const int idx = tid + i * 256;
        const int br  = idx >> 2;
        const int bc  = (idx & 3) * 8;
        uint4 vb = make_uint4(0u, 0u, 0u, 0u);
        if (n0 + br < N)
          vb = *(const uint4*)(W + (size_t)(n0 + br) * K + k0 + bc);
        *(uint4*)(&sB[0][br * LDB_S + bc]) = vb;
      }
      __syncthreads();
      frag_mma(&sA[0][0], &sB[0][0], wm, wn, lm, hh, acc);
      __syncthreads();
    }
  }

  // ---- epilogue: bias + activation + stores (C layout: VGPR r -> M = r + 8*hh) ----
#pragma unroll
  for (int mi = 0; mi < 2; ++mi) {
#pragma unroll
    for (int ni = 0; ni < 2; ++ni) {
      const int n = n0 + wn * 32 + ni * 16 + lm;
      if (n < N) {
        const float bv = bias ? bias[n] : 0.f;
#pragma unroll
        for (int r = 0; r < 8; ++r) {
          const int m = m0 + wm * 32 + mi * 16 + hh * 8 + r;
          float val = acc[mi][ni][r] + bv;
          if (act) val = fmaxf(val, 0.f);
          if (C)   C[(size_t)m * ldc + n] = val;
          if (Cbf) Cbf[(size_t)m * N + n] = f2bf(val);
        }
      }
    }
  }
}

// ---------------- elementwise kernels ----------------
__global__ void cvt_bf16(const float* __restrict__ s, ushort_t* __restrict__ d, int n) {
  int i = blockIdx.x * blockDim.x + threadIdx.x;
  if (i < n) d[i] = f2bf(s[i]);
}

__global__ void fill_f32(float* p, float v, int n) {
  int i = blockIdx.x * blockDim.x + threadIdx.x;
  if (i < n) p[i] = v;
}
__global__ void fill_u16(ushort_t* p, ushort_t v, int n) {
  int i = blockIdx.x * blockDim.x + threadIdx.x;
  if (i < n) p[i] = v;
}

// gi0 at t==0: x0 is one-hot at index 41 -> gi0[b][i] = w_ih0[i][41] + b_ih0[i]
__global__ void gi0_init(const float* __restrict__ wih0, const float* __restrict__ bih0,
                         float* __restrict__ gi0, int Bt, int G, int Vd) {
  int idx = blockIdx.x * blockDim.x + threadIdx.x;
  if (idx >= Bt * G) return;
  int i = idx % G;
  gi0[idx] = wih0[(size_t)i * Vd + 41] + bih0[i];
}

// Wf = w_ih0[G,V] @ d_fc2w[V,H]  -> bf16 [G,H] (stored [N=G, K=H] row-major)
__global__ void fuse_w(const float* __restrict__ wih0, const float* __restrict__ d2w,
                       ushort_t* __restrict__ Wf, int G, int Hd, int Vd) {
  int idx = blockIdx.x * blockDim.x + threadIdx.x;
  if (idx >= G * Hd) return;
  int i = idx / Hd, j = idx % Hd;
  float s = 0.f;
  for (int v = 0; v < Vd; ++v) s += wih0[(size_t)i * Vd + v] * d2w[(size_t)v * Hd + j];
  Wf[idx] = f2bf(s);
}
// b_f = w_ih0 @ d_fc2b + b_ih0
__global__ void fuse_b(const float* __restrict__ wih0, const float* __restrict__ bih0,
                       const float* __restrict__ d2b, float* __restrict__ bf, int G, int Vd) {
  int i = blockIdx.x * blockDim.x + threadIdx.x;
  if (i >= G) return;
  float s = bih0[i];
  for (int v = 0; v < Vd; ++v) s += wih0[(size_t)i * Vd + v] * d2b[v];
  bf[i] = s;
}

// GRU gate combine: h = (1-z)*tanh(in + r*hn) + z*h_prev ; writes f32 state + bf16 copy
__global__ void gru_combine(const float* __restrict__ gi, const float* __restrict__ gh,
                            float* __restrict__ h, ushort_t* __restrict__ hbf,
                            int Bt, int Hd) {
  int idx = blockIdx.x * blockDim.x + threadIdx.x;
  if (idx >= Bt * Hd) return;
  int b = idx / Hd, j = idx % Hd;
  const float* gib = gi + (size_t)b * 3 * Hd;
  const float* ghb = gh + (size_t)b * 3 * Hd;
  float r = sigm(gib[j] + ghb[j]);
  float z = sigm(gib[Hd + j] + ghb[Hd + j]);
  float nn = tanhf(gib[2 * Hd + j] + r * ghb[2 * Hd + j]);
  float hn = (1.f - z) * nn + z * h[idx];
  h[idx]   = hn;
  hbf[idx] = f2bf(hn);
}

// KLD partials (deterministic two-stage reduction) + reparameterized latent (bf16)
__global__ void latent_kld(const float* __restrict__ mu, const float* __restrict__ lv,
                           const float* __restrict__ eps, ushort_t* __restrict__ encbf,
                           float* __restrict__ partial, int n) {
  __shared__ float red[256];
  int i = blockIdx.x * blockDim.x + threadIdx.x;
  float term = 0.f;
  if (i < n) {
    float m = mu[i], l = lv[i];
    encbf[i] = f2bf(eps[i] * expf(0.5f * l) + m);
    term = 1.f + l - m * m - expf(l);
  }
  red[threadIdx.x] = term;
  __syncthreads();
  for (int s = 128; s > 0; s >>= 1) {
    if (threadIdx.x < s) red[threadIdx.x] += red[threadIdx.x + s];
    __syncthreads();
  }
  if (threadIdx.x == 0) partial[blockIdx.x] = red[0];
}
__global__ void kld_finalize(const float* __restrict__ partial, int nb, float scale,
                             float* __restrict__ out) {
  if (threadIdx.x == 0) {
    float s = 0.f;
    for (int i = 0; i < nb; ++i) s += partial[i];
    *out = s * scale;
  }
}

// ---------------- host orchestration ----------------
extern "C" void kernel_launch(void* const* d_in, const int* in_sizes, int n_in,
                              void* d_out, int out_size, void* d_ws, size_t ws_size,
                              hipStream_t stream) {
  (void)in_sizes; (void)n_in; (void)out_size; (void)ws_size;

  const float* X    = (const float*)d_in[0];
  const float* eps  = (const float*)d_in[2];
  const float* e1w  = (const float*)d_in[3];  const float* e1b  = (const float*)d_in[4];
  const float* e2w  = (const float*)d_in[5];  const float* e2b  = (const float*)d_in[6];
  const float* e3w  = (const float*)d_in[7];  const float* e3b  = (const float*)d_in[8];
  const float* e41w = (const float*)d_in[9];  const float* e41b = (const float*)d_in[10];
  const float* e42w = (const float*)d_in[11]; const float* e42b = (const float*)d_in[12];
  const float* d1w  = (const float*)d_in[13]; const float* d1b  = (const float*)d_in[14];
  const float* d2w  = (const float*)d_in[15]; const float* d2b  = (const float*)d_in[16];
  const float* wih0 = (const float*)d_in[17]; const float* bih0 = (const float*)d_in[18];
  const float* whh0 = (const float*)d_in[19]; const float* bhh0 = (const float*)d_in[20];
  const float* wih1 = (const float*)d_in[21]; const float* bih1 = (const float*)d_in[22];
  const float* whh1 = (const float*)d_in[23]; const float* bhh1 = (const float*)d_in[24];

  float* decoded = (float*)d_out;                       // [B, SEQ, V]
  float* kld     = decoded + (size_t)B_ * SEQ_ * V_;    // scalar

  // workspace bump allocator (deterministic layout)
  char* p = (char*)d_ws;
  auto alloc = [&](size_t bytes) -> char* {
    char* r = p; p += (bytes + 255) & ~(size_t)255; return r;
  };

  ushort_t* Xbf    = (ushort_t*)alloc((size_t)B_ * FP_ * 2);
  ushort_t* w1bf   = (ushort_t*)alloc((size_t)2048 * 4096 * 2);
  ushort_t* w2bf   = (ushort_t*)alloc((size_t)1024 * 2048 * 2);
  ushort_t* w3bf   = (ushort_t*)alloc((size_t)512 * 1024 * 2);
  ushort_t* w41bf  = (ushort_t*)alloc((size_t)ENC_ * 512 * 2);
  ushort_t* w42bf  = (ushort_t*)alloc((size_t)ENC_ * 512 * 2);
  ushort_t* d1wbf  = (ushort_t*)alloc((size_t)H_ * ENC_ * 2);
  ushort_t* d2wbf  = (ushort_t*)alloc((size_t)V_ * H_ * 2);
  ushort_t* whh0bf = (ushort_t*)alloc((size_t)G_ * H_ * 2);
  ushort_t* wih1bf = (ushort_t*)alloc((size_t)G_ * H_ * 2);
  ushort_t* whh1bf = (ushort_t*)alloc((size_t)G_ * H_ * 2);
  ushort_t* Wfbf   = (ushort_t*)alloc((size_t)G_ * H_ * 2);
  float*    bfused = (float*)alloc((size_t)G_ * 4);

  ushort_t* h1abf  = (ushort_t*)alloc((size_t)B_ * 2048 * 2);
  ushort_t* h2bf   = (ushort_t*)alloc((size_t)B_ * 1024 * 2);
  ushort_t* h3bf   = (ushort_t*)alloc((size_t)B_ * 512 * 2);
  float*    muf    = (float*)alloc((size_t)B_ * ENC_ * 4);
  float*    lvf    = (float*)alloc((size_t)B_ * ENC_ * 4);
  ushort_t* encbf  = (ushort_t*)alloc((size_t)B_ * ENC_ * 2);
  float*    part   = (float*)alloc(128 * 4);
  float*    h0f    = (float*)alloc((size_t)B_ * H_ * 4);
  ushort_t* h0bf   = (ushort_t*)alloc((size_t)B_ * H_ * 2);
  float*    h1f    = (float*)alloc((size_t)B_ * H_ * 4);
  ushort_t* h1bf   = (ushort_t*)alloc((size_t)B_ * H_ * 2);
  float*    gi0    = (float*)alloc((size_t)B_ * G_ * 4);
  float*    gh0    = (float*)alloc((size_t)B_ * G_ * 4);
  float*    gi1    = (float*)alloc((size_t)B_ * G_ * 4);
  float*    gh1    = (float*)alloc((size_t)B_ * G_ * 4);

  auto cvt = [&](const float* s, ushort_t* d, size_t n) {
    cvt_bf16<<<(int)((n + 255) / 256), 256, 0, stream>>>(s, d, (int)n);
  };
  auto gemm = [&](const ushort_t* A, const ushort_t* Wt, const float* bias,
                  float* C, ushort_t* Cbf, int M, int N, int K, int ldc, int act) {
    dim3 grid((N + BN - 1) / BN, M / BM);
    gemm_bf16_wmma<<<grid, 256, 0, stream>>>(A, Wt, bias, C, Cbf, M, N, K, ldc, act);
  };

  // ---- precision conversion (per launch; deterministic) ----
  cvt(X,    Xbf,    (size_t)B_ * FP_);
  cvt(e1w,  w1bf,   (size_t)2048 * 4096);
  cvt(e2w,  w2bf,   (size_t)1024 * 2048);
  cvt(e3w,  w3bf,   (size_t)512 * 1024);
  cvt(e41w, w41bf,  (size_t)ENC_ * 512);
  cvt(e42w, w42bf,  (size_t)ENC_ * 512);
  cvt(d1w,  d1wbf,  (size_t)H_ * ENC_);
  cvt(d2w,  d2wbf,  (size_t)V_ * H_);
  cvt(whh0, whh0bf, (size_t)G_ * H_);
  cvt(wih1, wih1bf, (size_t)G_ * H_);
  cvt(whh1, whh1bf, (size_t)G_ * H_);

  // Fold the V=42 vocab hop out of the recurrence: Wf = w_ih0 @ d_fc2w, b_f = w_ih0@d_fc2b + b_ih0
  fuse_w<<<(G_ * H_ + 255) / 256, 256, 0, stream>>>(wih0, d2w, Wfbf, G_, H_, V_);
  fuse_b<<<(G_ + 255) / 256, 256, 0, stream>>>(wih0, bih0, d2b, bfused, G_, V_);

  // ---- VAE encoder ----
  gemm(Xbf,   w1bf,  e1b,  nullptr, h1abf, B_, 2048, FP_,  2048, 1);
  gemm(h1abf, w2bf,  e2b,  nullptr, h2bf,  B_, 1024, 2048, 1024, 1);
  gemm(h2bf,  w3bf,  e3b,  nullptr, h3bf,  B_, 512,  1024, 512,  1);
  gemm(h3bf,  w41bf, e41b, muf,     nullptr, B_, ENC_, 512, ENC_, 0);
  gemm(h3bf,  w42bf, e42b, lvf,     nullptr, B_, ENC_, 512, ENC_, 0);
  latent_kld<<<128, 256, 0, stream>>>(muf, lvf, eps, encbf, part, B_ * ENC_);
  kld_finalize<<<1, 32, 0, stream>>>(part, 128, -0.5f / (float)B_, kld);

  // lat = encoded @ d_fc1w^T + b -> h0 initial; h1 initial = 0
  gemm(encbf, d1wbf, d1b, h0f, h0bf, B_, H_, ENC_, H_, 0);
  fill_f32<<<(B_ * H_ + 255) / 256, 256, 0, stream>>>(h1f, 0.f, B_ * H_);
  fill_u16<<<(B_ * H_ + 255) / 256, 256, 0, stream>>>(h1bf, (ushort_t)0, B_ * H_);

  // ---- GRU decoder: 128 autoregressive steps (stream-ordered launches) ----
  const int cblk = (B_ * H_ + 255) / 256;
  for (int t = 0; t < SEQ_; ++t) {
    if (t == 0)
      gi0_init<<<(B_ * G_ + 255) / 256, 256, 0, stream>>>(wih0, bih0, gi0, B_, G_, V_);
    else
      gemm(h1bf, Wfbf, bfused, gi0, nullptr, B_, G_, H_, G_, 0);   // fused x-path
    gemm(h0bf, whh0bf, bhh0, gh0, nullptr, B_, G_, H_, G_, 0);
    gru_combine<<<cblk, 256, 0, stream>>>(gi0, gh0, h0f, h0bf, B_, H_);
    gemm(h0bf, wih1bf, bih1, gi1, nullptr, B_, G_, H_, G_, 0);
    gemm(h1bf, whh1bf, bhh1, gh1, nullptr, B_, G_, H_, G_, 0);     // reads prev h1
    gru_combine<<<cblk, 256, 0, stream>>>(gi1, gh1, h1f, h1bf, B_, H_);
    // decoded[:, t, :] = h1 @ d_fc2w^T + d_fc2b  (row stride SEQ*V)
    gemm(h1bf, d2wbf, d2b, decoded + (size_t)t * V_, nullptr, B_, V_, H_, SEQ_ * V_, 0);
  }
}
